// PositionalDependentLayer_26156350832796
// MI455X (gfx1250) — matrix operand order repaired and checked by reading, not compiled, not run on hardware
//
#include <hip/hip_runtime.h>
#include <hip/hip_bf16.h>
#include <stdint.h>

#define B_TOK        131072
#define CIN          256
#define COUT         256
#define NEXP         16
#define ROWS_PER_BLK 128
#define BLKS_PER_EXP 1024   // worst case: all tokens in one expert
#define LDS_STRIDE   40     // 32 K-elems padded to 40 ushorts (80B) -> conflict-free b128 reads

typedef __attribute__((ext_vector_type(16))) __bf16 v16bf;
typedef __attribute__((ext_vector_type(8)))  float  v8f;

// ---------- fp32 <-> bf16 split helpers ----------
__device__ __forceinline__ unsigned short f2bf(float f) {
    unsigned u = __builtin_bit_cast(unsigned, f);
    unsigned r = u + 0x7FFFu + ((u >> 16) & 1u);   // RNE to bf16
    return (unsigned short)(r >> 16);
}
__device__ __forceinline__ float bf2f(unsigned short h) {
    return __builtin_bit_cast(float, ((unsigned)h) << 16);
}

__device__ __forceinline__ int tile_of(float c0, float c1) {
    // af = c*2 + 0.5 ; xi = floor(af.x) % 4 ; yi = floor(af.y) % 4 ; id = 4*xi+yi
    int xi = ((int)floorf(fmaf(c0, 2.0f, 0.5f))) & 3;
    int yi = ((int)floorf(fmaf(c1, 2.0f, 0.5f))) & 3;
    return (xi << 2) | yi;
}

// ---------- K0: zero per-expert counters ----------
__global__ void k_zero(int* count, int* cursor) {
    int t = threadIdx.x;
    if (t < NEXP) { count[t] = 0; cursor[t] = 0; }
}

// ---------- K1: histogram + copy coords to output tail ----------
__global__ void k_count(const float* __restrict__ coords, int* __restrict__ count,
                        float* __restrict__ outCoords) {
    int i = blockIdx.x * 256 + threadIdx.x;
    if (i >= B_TOK) return;
    float c0 = coords[2 * i], c1 = coords[2 * i + 1];
    atomicAdd(&count[tile_of(c0, c1)], 1);
    outCoords[2 * i]     = c0;
    outCoords[2 * i + 1] = c1;
}

// ---------- K2: 16-entry exclusive scan ----------
__global__ void k_scan(const int* __restrict__ count, int* __restrict__ offs,
                       int* __restrict__ cursor) {
    if (threadIdx.x == 0) {
        int s = 0;
        for (int e = 0; e < NEXP; ++e) { offs[e] = s; cursor[e] = s; s += count[e]; }
    }
}

// ---------- K3: scatter token ids into exact buckets ----------
__global__ void k_scatter(const float* __restrict__ coords, int* __restrict__ cursor,
                          int* __restrict__ idxArr) {
    int i = blockIdx.x * 256 + threadIdx.x;
    if (i >= B_TOK) return;
    int e = tile_of(coords[2 * i], coords[2 * i + 1]);
    int slot = atomicAdd(&cursor[e], 1);
    idxArr[slot] = i;
}

// ---------- K4: split W fp32 -> bf16 hi/lo ----------
__global__ void k_wsplit(const float* __restrict__ W, unsigned short* __restrict__ Whi,
                         unsigned short* __restrict__ Wlo) {
    int i = blockIdx.x * 256 + threadIdx.x;
    if (i >= NEXP * COUT * CIN) return;
    float w = W[i];
    unsigned short h = f2bf(w);
    Whi[i] = h;
    Wlo[i] = f2bf(w - bf2f(h));
}

// ---------- K5: grouped GEMM, bf16x3 split via V_WMMA_F32_16X16X32_BF16 ----------
__global__ __launch_bounds__(256)
void k_gemm(const float* __restrict__ feats, const int* __restrict__ idxArr,
            const int* __restrict__ cntArr, const int* __restrict__ offsArr,
            const unsigned short* __restrict__ Whi, const unsigned short* __restrict__ Wlo,
            const float* __restrict__ bias, float* __restrict__ out) {
    const int tid  = threadIdx.x;
    const int lane = tid & 31;
    const int wave = tid >> 5;
    const int e    = blockIdx.x >> 10;              // expert
    const int blk  = blockIdx.x & (BLKS_PER_EXP - 1);

    const int cnt   = cntArr[e];
    const int brow0 = blk * ROWS_PER_BLK;
    if (brow0 >= cnt) return;                       // over-launched block: cheap exit
    const int offs  = offsArr[e];

    __shared__ unsigned short sBh[COUT * LDS_STRIDE];   // 20 KB
    __shared__ unsigned short sBl[COUT * LDS_STRIDE];   // 20 KB
    __shared__ int sTok[8][16];

    // gather token ids for this wave's 16-row stripe (clamped; stores masked later)
    if (lane < 16) {
        int r  = brow0 + wave * 16 + lane;
        int rc = (r < cnt) ? r : (cnt - 1);
        sTok[wave][lane] = idxArr[offs + rc];
    }
    __syncthreads();

    const int mrow = lane & 15;       // A row / C col within tile
    const int kgrp = lane >> 4;       // lane half -> K group
    const int goff = kgrp * 8;        // A K-offset per ISA 16-bit A layout

    const int    tA = sTok[wave][mrow];
    const float* ap = feats + (size_t)tA * CIN;

    v8f acc[16];
#pragma unroll
    for (int n = 0; n < 16; ++n) acc[n] = (v8f){0, 0, 0, 0, 0, 0, 0, 0};

    const unsigned short* whiB = Whi + (size_t)e * COUT * CIN;
    const unsigned short* wloB = Wlo + (size_t)e * COUT * CIN;

    for (int ks = 0; ks < 8; ++ks) {
        const int kb = ks * 32;
        __syncthreads();   // protect slice from previous iteration's readers
        // stage 256x32 hi/lo weight slice: [n][klocal], stride 40
        for (int q = tid; q < COUT * 4; q += 256) {
            int n = q >> 2, part = q & 3;
            uint4 dh = *(const uint4*)(whiB + (size_t)n * CIN + kb + part * 8);
            uint4 dl = *(const uint4*)(wloB + (size_t)n * CIN + kb + part * 8);
            *(uint4*)&sBh[n * LDS_STRIDE + part * 8] = dh;
            *(uint4*)&sBl[n * LDS_STRIDE + part * 8] = dl;
        }
        __syncthreads();

        // A fragment: lane covers K = kb+goff+0..7 (vgpr0-3) and kb+goff+16..23 (vgpr4-7)
        float4 p0 = *(const float4*)(ap + kb + goff);
        float4 p1 = *(const float4*)(ap + kb + goff + 4);
        float4 p2 = *(const float4*)(ap + kb + goff + 16);
        float4 p3 = *(const float4*)(ap + kb + goff + 20);
        float f[16] = {p0.x, p0.y, p0.z, p0.w, p1.x, p1.y, p1.z, p1.w,
                       p2.x, p2.y, p2.z, p2.w, p3.x, p3.y, p3.z, p3.w};
        union { v16bf v; unsigned short u[16]; } ah, al;
#pragma unroll
        for (int i = 0; i < 16; ++i) {
            unsigned short h = f2bf(f[i]);
            ah.u[i] = h;
            al.u[i] = f2bf(f[i] - bf2f(h));
        }

#pragma unroll
        for (int n = 0; n < 16; ++n) {
            const int col = n * 16 + mrow;     // B fragment: lane = column, K = 8*kgrp+0..7 and 8*kgrp+16..23
            const unsigned short* ph = &sBh[col * LDS_STRIDE + kgrp * 8];
            const unsigned short* pl = &sBl[col * LDS_STRIDE + kgrp * 8];
            union { v16bf v; uint4 q[2]; } bh, bl;
            bh.q[0] = *(const uint4*)ph;  bh.q[1] = *(const uint4*)(ph + 16);
            bl.q[0] = *(const uint4*)pl;  bl.q[1] = *(const uint4*)(pl + 16);
            acc[n] = __builtin_amdgcn_wmma_f32_16x16x32_bf16(false, ah.v, false, bh.v,
                                                             (short)0, acc[n], false, false);
            acc[n] = __builtin_amdgcn_wmma_f32_16x16x32_bf16(false, ah.v, false, bl.v,
                                                             (short)0, acc[n], false, false);
            acc[n] = __builtin_amdgcn_wmma_f32_16x16x32_bf16(false, al.v, false, bh.v,
                                                             (short)0, acc[n], false, false);
        }
    }

    // epilogue: C layout -> VGPR j holds M = j + 8*kgrp, N = mrow
#pragma unroll
    for (int n = 0; n < 16; ++n) {
        const int col = n * 16 + mrow;
        const float bv = bias[col];
#pragma unroll
        for (int j = 0; j < 8; ++j) {
            int m = j + kgrp * 8;
            int r = brow0 + wave * 16 + m;
            if (r < cnt) {
                float y = acc[n][j] + bv;
                y = (y >= 0.0f) ? y : 0.2f * y;   // LeakyReLU(0.2)
                out[(size_t)sTok[wave][m] * COUT + col] = y;
            }
        }
    }
}

extern "C" void kernel_launch(void* const* d_in, const int* in_sizes, int n_in,
                              void* d_out, int out_size, void* d_ws, size_t ws_size,
                              hipStream_t stream) {
    const float* in_feats  = (const float*)d_in[0];
    const float* in_coords = (const float*)d_in[1];
    const float* W         = (const float*)d_in[2];
    const float* bias      = (const float*)d_in[3];

    float* out       = (float*)d_out;
    float* outCoords = out + (size_t)B_TOK * COUT;   // tuple part 2

    char* ws = (char*)d_ws;
    int* count  = (int*)(ws + 0);
    int* cursor = (int*)(ws + 64);
    int* offs   = (int*)(ws + 128);
    int* idxArr = (int*)(ws + 1024);                         // B ints (512 KB)
    unsigned short* Whi = (unsigned short*)(ws + (1u << 20)); // 2 MB
    unsigned short* Wlo = (unsigned short*)(ws + (3u << 20)); // 2 MB

    k_zero<<<1, 32, 0, stream>>>(count, cursor);
    k_count<<<B_TOK / 256, 256, 0, stream>>>(in_coords, count, outCoords);
    k_scan<<<1, 32, 0, stream>>>(count, offs, cursor);
    k_scatter<<<B_TOK / 256, 256, 0, stream>>>(in_coords, cursor, idxArr);
    k_wsplit<<<(NEXP * COUT * CIN) / 256, 256, 0, stream>>>(W, Whi, Wlo);
    k_gemm<<<NEXP * BLKS_PER_EXP, 256, 0, stream>>>(in_feats, idxArr, count, offs,
                                                    Whi, Wlo, bias, out);
}
